// GLTCH_9655086482002
// MI455X (gfx1250) — matrix operand order repaired and checked by reading, not compiled, not run on hardware
//
#include <hip/hip_runtime.h>
#include <hip/hip_bf16.h>
#include <stdint.h>

typedef __attribute__((ext_vector_type(16))) _Float16 v16h;
typedef __attribute__((ext_vector_type(8)))  float    v8f;

#define CDIM 192
#define HDIM 6
#define HDD  32
#define TDIM 128
#define KMAX 768   // largest K seen by the GEMM (FF2)

// ---------------------------------------------------------------------------
// Embedding: x[b,t,c] = tok_emb[idx[b,t],c] + pos_emb[t,c]
// ---------------------------------------------------------------------------
__global__ void embed_kernel(const int* __restrict__ idx,
                             const float* __restrict__ tok,
                             const float* __restrict__ pos,
                             float* __restrict__ x,
                             int T, int C, size_t total) {
  size_t n = (size_t)blockIdx.x * blockDim.x + threadIdx.x;
  if (n >= total) return;
  int c = (int)(n % C);
  size_t row = n / C;
  int t = (int)(row % T);
  x[n] = tok[(size_t)idx[row] * C + c] + pos[(size_t)t * C + c];
}

// ---------------------------------------------------------------------------
// LayerNorm over last dim (C), writes f16 for WMMA consumption.
// ---------------------------------------------------------------------------
__global__ void ln_kernel(const float* __restrict__ x,
                          const float* __restrict__ w,
                          const float* __restrict__ b,
                          _Float16* __restrict__ out, int C) {
  int row = blockIdx.x;
  int t = threadIdx.x;
  const float* xr = x + (size_t)row * C;
  __shared__ float red[64];
  __shared__ float stat[2];
  float s = 0.f;
  for (int c = t; c < C; c += 64) s += xr[c];
  red[t] = s; __syncthreads();
  if (t == 0) { float tot = 0.f; for (int i = 0; i < 64; i++) tot += red[i]; stat[0] = tot / C; }
  __syncthreads();
  float mu = stat[0];
  float s2 = 0.f;
  for (int c = t; c < C; c += 64) { float d = xr[c] - mu; s2 += d * d; }
  __syncthreads();
  red[t] = s2; __syncthreads();
  if (t == 0) { float tot = 0.f; for (int i = 0; i < 64; i++) tot += red[i]; stat[1] = rsqrtf(tot / C + 1e-5f); }
  __syncthreads();
  float rs = stat[1];
  for (int c = t; c < C; c += 64)
    out[(size_t)row * C + c] = (_Float16)((xr[c] - mu) * rs * w[c] + b[c]);
}

// ---------------------------------------------------------------------------
// Pack a row-major f32 W[K,N] into WMMA-B fragment-tile order, f16:
//   packed[((nt*Kt + kt)*32 + lane)*16 + i] = W[kt*32 + 16*(lane>>4) + i][nt*16 + (lane&15)]
// so each lane's 16 halves are 32 contiguous bytes (2x global_load_b128).
// ---------------------------------------------------------------------------
__global__ void pack_w_kernel(const float* __restrict__ src,
                              _Float16* __restrict__ dst, int K, int N) {
  size_t p = (size_t)blockIdx.x * blockDim.x + threadIdx.x;
  size_t tot = (size_t)K * N;
  if (p >= tot) return;
  int i    = (int)(p & 15);
  int lane = (int)((p >> 4) & 31);
  size_t tile = p >> 9;
  int ktiles = K >> 5;
  int kt = (int)(tile % ktiles);
  int nt = (int)(tile / ktiles);
  int k = kt * 32 + 16 * (lane >> 4) + i;
  int n = nt * 16 + (lane & 15);
  dst[p] = (_Float16)src[(size_t)k * N + n];
}

// Same, but source is per-layer QKV weight [H, C, HD] with logical B[k=c][n=h*HD+d].
__global__ void pack_wqkv_kernel(const float* __restrict__ src,
                                 _Float16* __restrict__ dst, int K, int N, int HD) {
  size_t p = (size_t)blockIdx.x * blockDim.x + threadIdx.x;
  size_t tot = (size_t)K * N;
  if (p >= tot) return;
  int i    = (int)(p & 15);
  int lane = (int)((p >> 4) & 31);
  size_t tile = p >> 9;
  int ktiles = K >> 5;
  int kt = (int)(tile % ktiles);
  int nt = (int)(tile / ktiles);
  int k = kt * 32 + 16 * (lane >> 4) + i;
  int n = nt * 16 + (lane & 15);
  int h = n / HD, d = n % HD;
  dst[p] = (_Float16)src[((size_t)h * K + k) * HD + d];
}

// ---------------------------------------------------------------------------
// Pack attention activations to f16:
//   q16,k16: [B*H, T, HD]  (d-contiguous)   vT16: [B*H, HD, T] (s-contiguous)
// ---------------------------------------------------------------------------
__global__ void pack_attn_kernel(const float* __restrict__ q,
                                 const float* __restrict__ k,
                                 const float* __restrict__ v,
                                 _Float16* __restrict__ q16,
                                 _Float16* __restrict__ k16,
                                 _Float16* __restrict__ vT16,
                                 int H, int T, int HD, int C, size_t total) {
  size_t n = (size_t)blockIdx.x * blockDim.x + threadIdx.x;
  if (n >= total) return;
  int d = (int)(n % HD);
  size_t r = n / HD;
  int t = (int)(r % T);
  size_t bh = r / T;
  int h = (int)(bh % H);
  size_t b = bh / H;
  size_t src = ((size_t)b * T + t) * C + (size_t)h * HD + d;
  q16[n] = (_Float16)q[src];
  k16[n] = (_Float16)k[src];
  vT16[(bh * HD + d) * T + t] = (_Float16)v[src];
}

// ---------------------------------------------------------------------------
// f16 WMMA GEMM: D = act(A[M,K] x B[K,N] + bias) + resid
// Block = 4 waves sharing one 16-row A strip, staged into LDS with
// global_load_async_to_lds_b128 (ASYNCcnt).  Each wave owns NTW 16-col tiles,
// so one ds_load_b128 A-fragment feeds NTW WMMAs.  B is pre-packed so each
// lane's fragment is 32 contiguous bytes; next K-tile is prefetched.
// Template params prune the epilogue at compile time.
// ---------------------------------------------------------------------------
template<int NTW, bool BIAS, bool RESID, bool GELU, bool OUT32, bool OUT16>
__global__ void gemm_f16_kernel(const _Float16* __restrict__ A,
                                const _Float16* __restrict__ Bp,
                                const float* __restrict__ bias,
                                const float* __restrict__ resid,
                                float* __restrict__ Dout,
                                _Float16* __restrict__ D16,
                                int M, int N, int K) {
  __shared__ _Float16 sA[16 * KMAX];
  int tid  = threadIdx.x;
  int wave = tid >> 5;
  int lane = tid & 31;
  int h2 = lane >> 4;
  int rc = lane & 15;
  int m0 = blockIdx.y * 16;
  int nt0 = blockIdx.x * (4 * NTW) + wave * NTW;
  int ktiles = K >> 5;

  // ---- async-stage the contiguous 16xK A strip into LDS ----
  {
    const _Float16* Astrip = A + (size_t)m0 * K;
    unsigned ldsbase = (unsigned)(uintptr_t)(&sA[0]);
    int chunks = 2 * K;                 // 16-byte chunks in the strip
    for (int c = tid; c < chunks; c += 128) {
      const _Float16* gp = Astrip + (size_t)c * 8;
      unsigned lo = ldsbase + (unsigned)c * 16u;
      asm volatile("global_load_async_to_lds_b128 %0, %1, off"
                   :: "v"(lo), "v"(gp) : "memory");
    }
    asm volatile("s_wait_asynccnt 0x0" ::: "memory");
  }
  __syncthreads();

  v8f acc[NTW] = {};
  for (int kt = 0; kt < ktiles; kt++) {
    v16h a;
    const _Float16* ap = sA + (size_t)rc * K + kt * 32 + 16 * h2;
#pragma unroll
    for (int i = 0; i < 8; i++) a[i] = ap[i];
#pragma unroll
    for (int i = 0; i < 8; i++) a[8 + i] = ap[8 + i];
#pragma unroll
    for (int j = 0; j < NTW; j++) {
      const _Float16* bp = Bp + ((((size_t)(nt0 + j) * ktiles + kt) * 32) + lane) * 16;
      if (kt + 1 < ktiles) __builtin_prefetch(bp + 512, 0, 0);  // next K-tile
      v16h bfr;
#pragma unroll
      for (int i = 0; i < 16; i++) bfr[i] = bp[i];
      acc[j] = __builtin_amdgcn_wmma_f32_16x16x32_f16(false, a, false, bfr,
                                                      (short)0, acc[j], false, false);
    }
  }
#pragma unroll
  for (int j = 0; j < NTW; j++) {
    int n = (nt0 + j) * 16 + rc;
#pragma unroll
    for (int r = 0; r < 8; r++) {
      int m = m0 + r + 8 * h2;
      float v = acc[j][r];
      if (BIAS)  v += bias[n];
      if (GELU)  v = 0.5f * v * (1.0f + erff(v * 0.70710678118654752f));
      size_t o = (size_t)m * N + n;
      if (RESID) v += resid[o];
      if (OUT32) Dout[o] = v;
      if (OUT16) D16[o] = (_Float16)v;
    }
  }
}

// ---------------------------------------------------------------------------
// Attention for one (b,h): S = q k^T * scale (WMMA, K=HD=32, single step),
// causal softmax in 64KB LDS (P narrowed to f16 in place), O = P v (WMMA, K=128).
// 128 threads = 4 waves.  All operand fetches are 16/32-byte contiguous.
// ---------------------------------------------------------------------------
__global__ void attn_kernel(const _Float16* __restrict__ q16,
                            const _Float16* __restrict__ k16,
                            const _Float16* __restrict__ vT16,
                            _Float16* __restrict__ o16,
                            int H, int C) {
  int bh = blockIdx.x;
  int b_ = bh / H;
  int h_ = bh % H;
  int tid  = threadIdx.x;
  int wave = tid >> 5;
  int lane = tid & 31;
  int h2 = lane >> 4;
  int rc = lane & 15;
  __shared__ float sS[TDIM * TDIM];   // scores, later f16 P per-row in place

  const size_t qkbase = (size_t)bh * TDIM * HDD;  // [BH,T,HD]
  const size_t vbase  = (size_t)bh * HDD * TDIM;  // [BH,HD,T]
  const float scale = 0.17677669529663687f;       // 1/sqrt(32)

  for (int tile = wave; tile < 64; tile += 4) {
    int m0 = (tile >> 3) * 16;
    int n0 = (tile & 7) * 16;
    v16h a, bfr; v8f c = {};
    const _Float16* ap = q16 + qkbase + (size_t)(m0 + rc) * HDD + 16 * h2;
#pragma unroll
    for (int i = 0; i < 8; i++) a[i] = ap[i];
#pragma unroll
    for (int i = 0; i < 8; i++) a[8 + i] = ap[8 + i];
    const _Float16* bp = k16 + qkbase + (size_t)(n0 + rc) * HDD + 16 * h2;
#pragma unroll
    for (int i = 0; i < 16; i++) bfr[i] = bp[i];
    c = __builtin_amdgcn_wmma_f32_16x16x32_f16(false, a, false, bfr,
                                               (short)0, c, false, false);
#pragma unroll
    for (int r = 0; r < 8; r++)
      sS[(m0 + r + 8 * h2) * TDIM + n0 + rc] = c[r] * scale;
  }
  __syncthreads();

  // causal softmax, one thread per row; narrow P to f16 in place (safe: f16
  // slot s covers bytes [2s,2s+2) while f32 elem s sits at byte 4s, ascending).
  {
    int t = tid;
    float* rowp = sS + (size_t)t * TDIM;
    _Float16* rowh = (_Float16*)rowp;
    float mx = -3.4e38f;
    for (int s = 0; s <= t; s++) mx = fmaxf(mx, rowp[s]);
    float sum = 0.f;
    for (int s = 0; s < TDIM; s++) {
      float e = (s <= t) ? __expf(rowp[s] - mx) : 0.f;
      sum += e;
      rowh[s] = (_Float16)e;
    }
    float inv = 1.f / sum;
    for (int s = 0; s < TDIM; s++) rowh[s] = (_Float16)((float)rowh[s] * inv);
  }
  __syncthreads();

  for (int tile = wave; tile < 16; tile += 4) {
    int m0 = (tile >> 1) * 16;
    int n0 = (tile & 1) * 16;
    v8f c = {};
    for (int k0 = 0; k0 < TDIM; k0 += 32) {
      v16h a, bfr;
      const _Float16* ap = (const _Float16*)(sS + (size_t)(m0 + rc) * TDIM) + k0 + 16 * h2;
#pragma unroll
      for (int i = 0; i < 8; i++) a[i] = ap[i];
#pragma unroll
      for (int i = 0; i < 8; i++) a[8 + i] = ap[8 + i];
      const _Float16* bp = vT16 + vbase + (size_t)(n0 + rc) * TDIM + k0 + 16 * h2;
#pragma unroll
      for (int i = 0; i < 16; i++) bfr[i] = bp[i];
      c = __builtin_amdgcn_wmma_f32_16x16x32_f16(false, a, false, bfr,
                                                 (short)0, c, false, false);
    }
#pragma unroll
    for (int r = 0; r < 8; r++)
      o16[((size_t)b_ * TDIM + m0 + r + 8 * h2) * C + h_ * HDD + n0 + rc] =
          (_Float16)c[r];
  }
}

// ---------------------------------------------------------------------------
// Loss: per-row log-softmax at target, mean-reduced via atomicAdd.
// ---------------------------------------------------------------------------
__global__ void zero_loss_kernel(float* p) { *p = 0.f; }

__global__ void loss_kernel(const float* __restrict__ logits,
                            const int* __restrict__ targets,
                            float* __restrict__ loss, int V, float invBT) {
  int row = blockIdx.x;
  const float* lr = logits + (size_t)row * V;
  __shared__ float red[256];
  int t = threadIdx.x;
  float mx = -3.4e38f;
  for (int c = t; c < V; c += 256) mx = fmaxf(mx, lr[c]);
  red[t] = mx; __syncthreads();
  if (t == 0) { float m = red[0]; for (int i = 1; i < 256; i++) m = fmaxf(m, red[i]); red[0] = m; }
  __syncthreads();
  mx = red[0];
  __syncthreads();
  float s = 0.f;
  for (int c = t; c < V; c += 256) s += __expf(lr[c] - mx);
  red[t] = s; __syncthreads();
  if (t == 0) {
    float tot = 0.f; for (int i = 0; i < 256; i++) tot += red[i];
    float lp = lr[targets[row]] - mx - __logf(tot);
    atomicAdd(loss, -lp * invBT);
  }
}

// ---------------------------------------------------------------------------
// Host orchestration
// ---------------------------------------------------------------------------
extern "C" void kernel_launch(void* const* d_in, const int* in_sizes, int n_in,
                              void* d_out, int out_size, void* d_ws, size_t ws_size,
                              hipStream_t stream) {
  (void)in_sizes; (void)n_in; (void)out_size; (void)ws_size;
  constexpr int V = 32000, C = CDIM, H = HDIM, HD = HDD, L = 6, B = 64, T = TDIM;
  constexpr int BT = B * T, FF = 4 * C;

  const int*   idx     = (const int*)d_in[0];
  const int*   targets = (const int*)d_in[1];
  const float* tok_emb = (const float*)d_in[2];
  const float* pos_emb = (const float*)d_in[3];
  const float* ln1_w   = (const float*)d_in[4];
  const float* ln1_b   = (const float*)d_in[5];
  const float* wq      = (const float*)d_in[6];
  const float* wk      = (const float*)d_in[7];
  const float* wv      = (const float*)d_in[8];
  const float* proj_w  = (const float*)d_in[9];
  const float* proj_b  = (const float*)d_in[10];
  const float* ln2_w   = (const float*)d_in[11];
  const float* ln2_b   = (const float*)d_in[12];
  const float* ff1_w   = (const float*)d_in[13];
  const float* ff1_b   = (const float*)d_in[14];
  const float* ff2_w   = (const float*)d_in[15];
  const float* ff2_b   = (const float*)d_in[16];
  const float* lnf_w   = (const float*)d_in[17];
  const float* lnf_b   = (const float*)d_in[18];
  const float* head_w  = (const float*)d_in[19];
  const float* head_b  = (const float*)d_in[20];

  uint8_t* base = (uint8_t*)d_ws;
  size_t off = 0;
  auto alloc = [&](size_t bytes) -> void* {
    off = (off + 255) & ~(size_t)255;
    void* p = base + off;
    off += bytes;
    return p;
  };

  float*     x     = (float*)    alloc((size_t)BT * C * 4);
  _Float16*  h16   = (_Float16*) alloc((size_t)BT * C * 2);
  float*     q32   = (float*)    alloc((size_t)BT * C * 4);
  float*     k32   = (float*)    alloc((size_t)BT * C * 4);
  float*     v32   = (float*)    alloc((size_t)BT * C * 4);
  _Float16*  q16   = (_Float16*) alloc((size_t)BT * C * 2);
  _Float16*  k16   = (_Float16*) alloc((size_t)BT * C * 2);
  _Float16*  vT16  = (_Float16*) alloc((size_t)BT * C * 2);
  _Float16*  o16   = (_Float16*) alloc((size_t)BT * C * 2);
  _Float16*  ff16  = (_Float16*) alloc((size_t)BT * FF * 2);
  _Float16*  wq16  = (_Float16*) alloc((size_t)C * C * 2);
  _Float16*  wk16  = (_Float16*) alloc((size_t)C * C * 2);
  _Float16*  wv16  = (_Float16*) alloc((size_t)C * C * 2);
  _Float16*  pw16  = (_Float16*) alloc((size_t)C * C * 2);
  _Float16*  f116  = (_Float16*) alloc((size_t)C * FF * 2);
  _Float16*  f216  = (_Float16*) alloc((size_t)FF * C * 2);
  _Float16*  hw16  = (_Float16*) alloc((size_t)C * V * 2);

  float* logits = (float*)d_out;
  float* lossp  = (float*)d_out + (size_t)BT * V;

  {
    size_t total = (size_t)BT * C;
    embed_kernel<<<(unsigned)((total + 255) / 256), 256, 0, stream>>>(
        idx, tok_emb, pos_emb, x, T, C, total);
  }

  for (int l = 0; l < L; l++) {
    ln_kernel<<<BT, 64, 0, stream>>>(x, ln1_w + (size_t)l * C, ln1_b + (size_t)l * C, h16, C);

    int qkvTot = H * C * HD;
    pack_wqkv_kernel<<<(qkvTot + 255) / 256, 256, 0, stream>>>(wq + (size_t)l * qkvTot, wq16, C, C, HD);
    pack_wqkv_kernel<<<(qkvTot + 255) / 256, 256, 0, stream>>>(wk + (size_t)l * qkvTot, wk16, C, C, HD);
    pack_wqkv_kernel<<<(qkvTot + 255) / 256, 256, 0, stream>>>(wv + (size_t)l * qkvTot, wv16, C, C, HD);

    // QKV GEMMs: [BT,C] x [C,C] -> f32.  N=192 = 4 waves * 3 tiles -> grid.x=1
    dim3 gC(1, BT / 16);
    gemm_f16_kernel<3, false, false, false, true, false><<<gC, 128, 0, stream>>>(
        h16, wq16, nullptr, nullptr, q32, nullptr, BT, C, C);
    gemm_f16_kernel<3, false, false, false, true, false><<<gC, 128, 0, stream>>>(
        h16, wk16, nullptr, nullptr, k32, nullptr, BT, C, C);
    gemm_f16_kernel<3, false, false, false, true, false><<<gC, 128, 0, stream>>>(
        h16, wv16, nullptr, nullptr, v32, nullptr, BT, C, C);

    {
      size_t total = (size_t)BT * C;
      pack_attn_kernel<<<(unsigned)((total + 255) / 256), 256, 0, stream>>>(
          q32, k32, v32, q16, k16, vT16, H, T, HD, C, total);
    }

    attn_kernel<<<B * H, 128, 0, stream>>>(q16, k16, vT16, o16, H, C);

    // proj: x = x + o @ proj_w + proj_b
    pack_w_kernel<<<(int)(((size_t)C * C + 255) / 256), 256, 0, stream>>>(proj_w + (size_t)l * C * C, pw16, C, C);
    gemm_f16_kernel<3, true, true, false, true, false><<<gC, 128, 0, stream>>>(
        o16, pw16, proj_b + (size_t)l * C, x, x, nullptr, BT, C, C);

    ln_kernel<<<BT, 64, 0, stream>>>(x, ln2_w + (size_t)l * C, ln2_b + (size_t)l * C, h16, C);

    // FF1: gelu(h @ ff1_w + ff1_b) -> f16 only.  N=768 -> grid.x=4
    pack_w_kernel<<<(int)(((size_t)C * FF + 255) / 256), 256, 0, stream>>>(ff1_w + (size_t)l * C * FF, f116, C, FF);
    dim3 gFF1(FF / 192, BT / 16);
    gemm_f16_kernel<3, true, false, true, false, true><<<gFF1, 128, 0, stream>>>(
        h16, f116, ff1_b + (size_t)l * FF, nullptr, nullptr, ff16, BT, FF, C);

    // FF2: x = x + f @ ff2_w + ff2_b  (K=768)
    pack_w_kernel<<<(int)(((size_t)FF * C + 255) / 256), 256, 0, stream>>>(ff2_w + (size_t)l * FF * C, f216, FF, C);
    gemm_f16_kernel<3, true, true, false, true, false><<<gC, 128, 0, stream>>>(
        ff16, f216, ff2_b + (size_t)l * C, x, x, nullptr, BT, C, FF);
  }

  // final LN + head GEMM straight into d_out.  N=32000 = 100 blocks * 320 cols
  ln_kernel<<<BT, 64, 0, stream>>>(x, lnf_w, lnf_b, h16, C);
  pack_w_kernel<<<(int)(((size_t)C * V + 255) / 256), 256, 0, stream>>>(head_w, hw16, C, V);
  dim3 gHead(V / 320, BT / 16);
  gemm_f16_kernel<5, true, false, false, true, false><<<gHead, 128, 0, stream>>>(
      h16, hw16, head_b, nullptr, logits, nullptr, BT, V, C);

  zero_loss_kernel<<<1, 1, 0, stream>>>(lossp);
  loss_kernel<<<BT, 256, 0, stream>>>(logits, targets, lossp, V, 1.0f / BT);
}
